// ClassifierROI_87651692577475
// MI455X (gfx1250) — matrix operand order repaired and checked by reading, not compiled, not run on hardware
//
#include <hip/hip_runtime.h>

typedef __attribute__((ext_vector_type(16))) _Float16 v16h;
typedef __attribute__((ext_vector_type(8)))  float    v8f;

// ---------------------------------------------------------------------------
// WMMA helpers (CDNA5 wave32, V_WMMA_F32_16X16X32_F16)
// A 16x32 f16 layout: lane<16 row M=lane, half h -> K=(h<8?h:h+8); lane>=16: +8
// B 32x16 f16 layout: lane%16 = column N, lane/16 selects K-half, half h -> K
// C/D 16x16 f32:      lane%16 = N, vgpr r -> M = r + 8*(lane>=16)
// B panels are staged as [N][Kpad] so each lane's 16 halves are two contiguous 8-half runs
// (2x ds_load_b128). Row strides are padded to 4 (mod 8) dwords and 16B
// aligned: bank-conflict-free across a 16-lane group, still vectorizable.
// ---------------------------------------------------------------------------
__device__ __forceinline__ v8f wmma_f16(v16h a, v16h b, v8f c) {
  return __builtin_amdgcn_wmma_f32_16x16x32_f16(
      /*neg_a=*/false, a, /*neg_b=*/false, b,
      /*c_mod=*/(short)0, c, /*reuse_a=*/false, /*reuse_b=*/false);
}

__device__ __forceinline__ v16h lds_load_A(const _Float16* A, int mBase, int kBase, int ldk) {
  const int lane = threadIdx.x & 31;
  const int m = mBase + (lane & 15);
  const int koff = (lane >> 4) << 3;          // 0 or 8
  v16h a;
#pragma unroll
  for (int h = 0; h < 16; ++h) {
    int k = (h < 8 ? h : h + 8) + koff;       // two contiguous 8-half runs
    a[h] = A[m * ldk + kBase + k];
  }
  return a;
}

// B stored as [N][ldk]: lane reads two contiguous 8-half runs at kBase+kgrp and kBase+16+kgrp
__device__ __forceinline__ v16h lds_load_B(const _Float16* B, int kBase, int nBase, int ldk) {
  const int lane = threadIdx.x & 31;
  const int n = nBase + (lane & 15);
  const int kgrp = (lane >> 4) << 3;          // 0 or 8
  v16h b;
#pragma unroll
  for (int h = 0; h < 16; ++h) {
    b[h] = B[n * ldk + kBase + kgrp + (h < 8 ? h : h + 8)];
  }
  return b;
}

// ---------------------------------------------------------------------------
// Kernel 1: per-ROI bilinear crop (32x32x3) + conv1 3x3 s2 pad1 (3->16) + ReLU
// Implicit GEMM: M=256 (16x16 outputs), N=16, K=27 padded to 32 (ldk 40).
// ---------------------------------------------------------------------------
__global__ __launch_bounds__(256) void k_crop_conv1(
    const float* __restrict__ img, const int* __restrict__ bbox,
    const float* __restrict__ W1, _Float16* __restrict__ x1)
{
  __shared__ float crop[3 * 34 * 34];                 // +1 halo each side
  __shared__ __align__(16) _Float16 As[256 * 40];
  __shared__ __align__(16) _Float16 Bs[16 * 40];
  __shared__ int g0y[32], g1y[32], g0x[32], g1x[32];
  __shared__ float wwy[32], wwx[32];

  const int roi = blockIdx.x;
  const int t   = threadIdx.x;
  const int b   = bbox[roi * 6 + 0];
  const int y0  = bbox[roi * 6 + 1];
  const int x0  = bbox[roi * 6 + 2];
  const int y1  = bbox[roi * 6 + 3];
  const int x1e = bbox[roi * 6 + 4];

  if (t < 64) {
    const int i  = t & 31;
    const int lo = (t < 32) ? y0 : x0;
    const int hi = (t < 32) ? y1 : x1e;
    const int sz = hi - lo;
    float s  = fmaxf(((float)i + 0.5f) * (float)sz / 32.0f - 0.5f, 0.0f);
    int   i0 = (int)floorf(s);
    int   i1 = min(i0 + 1, sz - 1);
    float w  = s - (float)i0;
    if (t < 32) { g0y[i] = lo + i0; g1y[i] = lo + i1; wwy[i] = w; }
    else        { g0x[i] = lo + i0; g1x[i] = lo + i1; wwx[i] = w; }
  }
  for (int i = t; i < 3 * 34 * 34; i += 256) crop[i] = 0.0f;
  __syncthreads();

  const size_t imgBase = (size_t)b * 3u * 1024u * 1024u;
  for (int i = t; i < 3 * 32 * 32; i += 256) {
    const int c = i >> 10, rem = i & 1023, oy = rem >> 5, ox = rem & 31;
    const float* ip = img + imgBase + (size_t)c * 1024u * 1024u;
    const float v00 = ip[g0y[oy] * 1024 + g0x[ox]];
    const float v01 = ip[g0y[oy] * 1024 + g1x[ox]];
    const float v10 = ip[g1y[oy] * 1024 + g0x[ox]];
    const float v11 = ip[g1y[oy] * 1024 + g1x[ox]];
    const float wy = wwy[oy], wx = wwx[ox];
    crop[(c * 34 + oy + 1) * 34 + ox + 1] =
        (1.0f - wy) * ((1.0f - wx) * v00 + wx * v01) +
        wy          * ((1.0f - wx) * v10 + wx * v11);
  }
  // B panel: W1 (16,3,3,3) -> Bs[n][k], k = c*9+dy*3+dx, pad k in [27,32)
  for (int i = t; i < 16 * 32; i += 256) {
    const int n = i >> 5, k = i & 31;
    Bs[n * 40 + k] = (_Float16)((k < 27) ? W1[n * 27 + k] : 0.0f);
  }
  __syncthreads();

  // im2col: one row m per thread
  {
    const int m = t, oy = m >> 4, ox = m & 15;
#pragma unroll
    for (int k = 0; k < 32; ++k) {
      float v = 0.0f;
      if (k < 27) {
        const int c = k / 9, r = k % 9, dy = r / 3, dx = r % 3;
        v = crop[(c * 34 + 2 * oy + dy) * 34 + 2 * ox + dx];  // halo folded in
      }
      As[m * 40 + k] = (_Float16)v;
    }
  }
  __syncthreads();

  const int wave = t >> 5, lane = t & 31;
  _Float16* out = x1 + (size_t)roi * 16 * 256;   // [c16][m256], f16
#pragma unroll
  for (int tt = 0; tt < 2; ++tt) {
    const int mBase = wave * 32 + tt * 16;
    v16h a  = lds_load_A(As, mBase, 0, 40);
    v16h bm = lds_load_B(Bs, 0, 0, 40);
    v8f acc = {};
    acc = wmma_f16(a, bm, acc);
    const int n    = lane & 15;
    const int mOff = mBase + ((lane >> 4) << 3);
#pragma unroll
    for (int r = 0; r < 8; ++r)
      out[n * 256 + mOff + r] = (_Float16)fmaxf(acc[r], 0.0f);
  }
}

// ---------------------------------------------------------------------------
// Kernel 2: conv2 3x3 s2 pad1 (16->32). M=64, N=32, K=144 pad 160 (ldk 168).
// Also accumulates per-channel sum / sumsq for batch-norm.
// ---------------------------------------------------------------------------
__global__ __launch_bounds__(256) void k_conv2(
    const _Float16* __restrict__ x1, const float* __restrict__ W2,
    float* __restrict__ x2, float* __restrict__ gsum, float* __restrict__ gsq)
{
  __shared__ __align__(16) _Float16 in2[16 * 18 * 18];
  __shared__ __align__(16) _Float16 As[64 * 168];
  __shared__ __align__(16) _Float16 Bs[32 * 168];
  __shared__ float blkS[32], blkQ[32];

  const int roi = blockIdx.x;
  const int t   = threadIdx.x;
  for (int i = t; i < 16 * 18 * 18; i += 256) in2[i] = (_Float16)0.0f;
  if (t < 32) { blkS[t] = 0.0f; blkQ[t] = 0.0f; }
  __syncthreads();

  const _Float16* src = x1 + (size_t)roi * 4096;
  for (int i = t; i < 4096; i += 256) {
    const int c = i >> 8, rem = i & 255, y = rem >> 4, x = rem & 15;
    in2[(c * 18 + y + 1) * 18 + x + 1] = src[i];
  }
  // B: W2 (32,16,3,3) flat = [n][k(144)]; coalesced identity copy + zero pad
  for (int i = t; i < 32 * 144; i += 256) {
    const int n = i / 144, k = i % 144;
    Bs[n * 168 + k] = (_Float16)W2[i];
  }
  for (int i = t; i < 32 * 16; i += 256) {
    const int n = i >> 4, k = 144 + (i & 15);
    Bs[n * 168 + k] = (_Float16)0.0f;
  }
  __syncthreads();

  for (int i = t; i < 64 * 160; i += 256) {
    const int m = i / 160, k = i % 160;
    float v = 0.0f;
    if (k < 144) {
      const int c = k / 9, r = k % 9, dy = r / 3, dx = r % 3;
      const int oy = m >> 3, ox = m & 7;
      v = (float)in2[(c * 18 + 2 * oy + dy) * 18 + 2 * ox + dx];
    }
    As[m * 168 + k] = (_Float16)v;
  }
  __syncthreads();

  const int wave = t >> 5, lane = t & 31;
  const int mBase = (wave & 3) * 16, nBase = (wave >> 2) * 16;
  v8f acc = {};
#pragma unroll
  for (int kb = 0; kb < 5; ++kb) {
    v16h a  = lds_load_A(As, mBase, kb * 32, 168);
    v16h bm = lds_load_B(Bs, kb * 32, nBase, 168);
    acc = wmma_f16(a, bm, acc);
  }
  const int n    = nBase + (lane & 15);
  const int mOff = mBase + ((lane >> 4) << 3);
  float* dst = x2 + (size_t)roi * 2048;
  float s = 0.0f, q = 0.0f;
#pragma unroll
  for (int r = 0; r < 8; ++r) {
    const float v = acc[r];
    dst[n * 64 + mOff + r] = v;
    s += v; q += v * v;
  }
  s += __shfl_xor(s, 16);
  q += __shfl_xor(q, 16);
  if (lane < 16) { atomicAdd(&blkS[n], s); atomicAdd(&blkQ[n], q); }
  __syncthreads();
  if (t < 32) { atomicAdd(&gsum[t], blkS[t]); atomicAdd(&gsq[t], blkQ[t]); }
}

// ---------------------------------------------------------------------------
// Finalize BN stats -> scale/shift
// ---------------------------------------------------------------------------
__global__ void k_stats(const float* __restrict__ gsum, const float* __restrict__ gsq,
                        const float* __restrict__ g, const float* __restrict__ bb,
                        float* __restrict__ scale, float* __restrict__ shift,
                        int nch, float inv_n)
{
  const int t = threadIdx.x;
  if (t < nch) {
    const float mean = gsum[t] * inv_n;
    const float var  = gsq[t] * inv_n - mean * mean;
    const float sc   = g[t] * rsqrtf(var + 1e-5f);
    scale[t] = sc;
    shift[t] = bb[t] - mean * sc;
  }
}

// ---------------------------------------------------------------------------
// Kernel 3: BN2+ReLU, conv3 3x3 s2 pad1 (32->64). M=16, N=64, K=288 (ldk 296).
// ---------------------------------------------------------------------------
__global__ __launch_bounds__(256) void k_conv3(
    const float* __restrict__ x2, const float* __restrict__ W3,
    const float* __restrict__ scale2, const float* __restrict__ shift2,
    float* __restrict__ x3, float* __restrict__ gsum, float* __restrict__ gsq)
{
  __shared__ __align__(16) _Float16 in3[32 * 10 * 10];
  __shared__ __align__(16) _Float16 As[16 * 296];
  __shared__ __align__(16) _Float16 Bs[64 * 296];
  __shared__ float blkS[64], blkQ[64];

  const int roi = blockIdx.x;
  const int t   = threadIdx.x;
  __builtin_prefetch(W3 + t * 64, 0, 3);   // global_prefetch path (18432 f32)
  for (int i = t; i < 32 * 10 * 10; i += 256) in3[i] = (_Float16)0.0f;
  __syncthreads();

  const float* src = x2 + (size_t)roi * 2048;
  for (int i = t; i < 2048; i += 256) {
    const int c = i >> 6, rem = i & 63, y = rem >> 3, x = rem & 7;
    const float v = fmaxf(scale2[c] * src[i] + shift2[c], 0.0f);
    in3[(c * 10 + y + 1) * 10 + x + 1] = (_Float16)v;
  }
  // B: W3 (64,32,3,3) flat = [n][k(288)]; coalesced identity copy
  for (int i = t; i < 64 * 288; i += 256) {
    const int n = i / 288, k = i % 288;
    Bs[n * 296 + k] = (_Float16)W3[i];
  }
  __syncthreads();

  for (int i = t; i < 16 * 288; i += 256) {
    const int m = i / 288, k = i % 288;
    const int c = k / 9, r = k % 9, dy = r / 3, dx = r % 3;
    const int oy = m >> 2, ox = m & 3;
    As[m * 296 + k] = in3[(c * 10 + 2 * oy + dy) * 10 + 2 * ox + dx];
  }
  __syncthreads();

  const int wave = t >> 5, lane = t & 31;
  if (wave < 4) {
    const int nBase = wave * 16;
    v8f acc = {};
#pragma unroll
    for (int kb = 0; kb < 9; ++kb) {
      v16h a  = lds_load_A(As, 0, kb * 32, 296);
      v16h bm = lds_load_B(Bs, kb * 32, nBase, 296);
      acc = wmma_f16(a, bm, acc);
    }
    const int n    = nBase + (lane & 15);
    const int mOff = (lane >> 4) << 3;
    float* dst = x3 + (size_t)roi * 1024;
    float s = 0.0f, q = 0.0f;
#pragma unroll
    for (int r = 0; r < 8; ++r) {
      const float v = acc[r];
      dst[n * 16 + mOff + r] = v;
      s += v; q += v * v;
    }
    s += __shfl_xor(s, 16);
    q += __shfl_xor(q, 16);
    if (lane < 16) { blkS[n] = s; blkQ[n] = q; }
  }
  __syncthreads();
  if (t < 64) { atomicAdd(&gsum[t], blkS[t]); atomicAdd(&gsq[t], blkQ[t]); }
}

// ---------------------------------------------------------------------------
// Kernel 4: BN3+ReLU + global avg pool + 64->2 linear; also emit labels.
// ---------------------------------------------------------------------------
__global__ __launch_bounds__(256) void k_head(
    const float* __restrict__ x3, const float* __restrict__ scale3,
    const float* __restrict__ shift3, const float* __restrict__ Wc,
    const float* __restrict__ bc, const int* __restrict__ bbox,
    float* __restrict__ out)
{
  const int roi = blockIdx.x * 256 + threadIdx.x;
  if (roi >= 4096) return;
  const float* src = x3 + (size_t)roi * 1024;
  float a0 = bc[0], a1 = bc[1];
  for (int c = 0; c < 64; ++c) {
    const float sc = scale3[c], sh = shift3[c];
    float s = 0.0f;
#pragma unroll
    for (int m = 0; m < 16; ++m) s += fmaxf(sc * src[c * 16 + m] + sh, 0.0f);
    s *= (1.0f / 16.0f);
    a0 += Wc[c] * s;
    a1 += Wc[64 + c] * s;
  }
  out[4096 + roi * 2 + 0] = a0;
  out[4096 + roi * 2 + 1] = a1;
  out[roi] = (float)bbox[roi * 6 + 5];
}

__global__ void k_zero(float* __restrict__ p, int n) {
  const int i = blockIdx.x * 256 + threadIdx.x;
  if (i < n) p[i] = 0.0f;
}

// ---------------------------------------------------------------------------
extern "C" void kernel_launch(void* const* d_in, const int* in_sizes, int n_in,
                              void* d_out, int out_size, void* d_ws, size_t ws_size,
                              hipStream_t stream)
{
  const float* inputs = (const float*)d_in[0];
  const int*   bboxes = (const int*)  d_in[1];
  const float* W1     = (const float*)d_in[2];
  const float* W2     = (const float*)d_in[3];
  const float* g2     = (const float*)d_in[4];
  const float* b2     = (const float*)d_in[5];
  const float* W3     = (const float*)d_in[6];
  const float* g3     = (const float*)d_in[7];
  const float* b3     = (const float*)d_in[8];
  const float* Wc     = (const float*)d_in[9];
  const float* bc     = (const float*)d_in[10];

  char* ws = (char*)d_ws;
  _Float16* x1 = (_Float16*)(ws);                       // 4096*16*256  f16 = 32 MiB
  float*    x2 = (float*)(ws + 33554432u);              // 4096*32*64   f32 = 32 MiB
  float*    x3 = (float*)(ws + 67108864u);              // 4096*64*16   f32 = 16 MiB
  float* stats = (float*)(ws + 83886080u);              // 384 floats
  float* gsum2 = stats;        float* gsq2 = stats + 32;
  float* scale2 = stats + 64;  float* shift2 = stats + 96;
  float* gsum3 = stats + 128;  float* gsq3 = stats + 192;
  float* scale3 = stats + 256; float* shift3 = stats + 320;

  k_zero<<<2, 256, 0, stream>>>(stats, 384);
  k_crop_conv1<<<4096, 256, 0, stream>>>(inputs, bboxes, W1, x1);
  k_conv2<<<4096, 256, 0, stream>>>(x1, W2, x2, gsum2, gsq2);
  k_stats<<<1, 32, 0, stream>>>(gsum2, gsq2, g2, b2, scale2, shift2, 32,
                                1.0f / (4096.0f * 64.0f));
  k_conv3<<<4096, 256, 0, stream>>>(x2, W3, scale2, shift2, x3, gsum3, gsq3);
  k_stats<<<1, 64, 0, stream>>>(gsum3, gsq3, g3, b3, scale3, shift3, 64,
                                1.0f / (4096.0f * 16.0f));
  k_head<<<16, 256, 0, stream>>>(x3, scale3, shift3, Wc, bc, bboxes, (float*)d_out);
}
